// DisConv_40372692583118
// MI455X (gfx1250) — matrix operand order repaired and checked by reading, not compiled, not run on hardware
//
#include <hip/hip_runtime.h>
#include <hip/hip_bf16.h>

#define N 6144
#define IN_DIM 512
#define CH 4
#define KD 64
#define NW (N / 32) /* 192 adjacency words per row */

typedef _Float16 h16v __attribute__((ext_vector_type(16)));
typedef _Float16 h8v  __attribute__((ext_vector_type(8)));
typedef float    f8v  __attribute__((ext_vector_type(8)));

static __device__ __forceinline__ f8v wmma_f16(h16v a, h16v b, f8v c) {
  // D = A(16x32 f16) x B(32x16 f16) + C(16x16 f32)
  return __builtin_amdgcn_wmma_f32_16x16x32_f16(false, a, false, b, (short)0, c,
                                                false, false);
}

// ---------------------------------------------------------------------------
// Kernel 1: bit-pack the dense 0/1 adjacency (151 MB int32 -> 4.5 MB bits) in
// TRANSPOSED form adjT[word][n], so the iteration kernel can fetch the 16
// words of an (n-tile, m-step) as one contiguous wave-uniform scalar load.
// ---------------------------------------------------------------------------
__global__ __launch_bounds__(256) void pack_adj_kernel(const int* __restrict__ adj,
                                                       unsigned* __restrict__ bitsT) {
  int idx = blockIdx.x * 256 + threadIdx.x;
  if (idx >= N * NW) return;
  int n = idx / NW, w = idx - n * NW;  // reads coalesced along w within row n
  const int4* p = (const int4*)(adj + (size_t)n * N + (size_t)w * 32);
  unsigned m = 0;
#pragma unroll
  for (int q = 0; q < 8; ++q) {
    int4 v = p[q];
    m |= (v.x > 0 ? 1u : 0u) << (q * 4 + 0);
    m |= (v.y > 0 ? 1u : 0u) << (q * 4 + 1);
    m |= (v.z > 0 ? 1u : 0u) << (q * 4 + 2);
    m |= (v.w > 0 ? 1u : 0u) << (q * 4 + 3);
  }
  bitsT[(size_t)w * N + n] = m;  // transposed (one-time scattered store)
}

// ---------------------------------------------------------------------------
// Kernel 2: per-channel projection + row L2-norm.
// Block = 256 threads = (c,k); handles 16 rows of `features`, streams W once.
// Writes f32 master copy + f16 row-major + f16 transposed (k-major) copies.
// ---------------------------------------------------------------------------
__global__ __launch_bounds__(256) void project_kernel(
    const float* __restrict__ feat, const float* __restrict__ W,
    const float* __restrict__ b, float* __restrict__ f32,
    _Float16* __restrict__ f16, _Float16* __restrict__ fT) {
  extern __shared__ char smem[];
  float* ldsF = (float*)smem;                               // 16*512 f32
  float* proj = (float*)(smem + 16 * IN_DIM * 4);           // 4*16*64 f32
  float* nrm = (float*)(smem + 16 * IN_DIM * 4 + CH * 16 * KD * 4);  // 64 f32
  const int tid = threadIdx.x;
  const int n0 = blockIdx.x * 16;
  for (int i = tid; i < 16 * IN_DIM; i += 256)
    ldsF[i] = feat[(size_t)n0 * IN_DIM + i];
  __syncthreads();
  const int c = tid >> 6, k = tid & 63;
  float acc[16];
  const float bb = b[c * KD + k];
#pragma unroll
  for (int r = 0; r < 16; ++r) acc[r] = bb;
  for (int d = 0; d < IN_DIM; ++d) {
    float w = W[((size_t)c * IN_DIM + d) * KD + k];
#pragma unroll
    for (int r = 0; r < 16; ++r) acc[r] += ldsF[r * IN_DIM + d] * w;
  }
#pragma unroll
  for (int r = 0; r < 16; ++r) proj[(c * 16 + r) * KD + k] = acc[r];
  __syncthreads();
  if (tid < 64) {
    const int cc = tid >> 4, rr = tid & 15;
    float ss = 0.f;
    for (int kk = 0; kk < KD; ++kk) {
      float v = proj[(cc * 16 + rr) * KD + kk];
      ss += v * v;
    }
    nrm[tid] = fmaxf(sqrtf(ss), 1e-12f);
  }
  __syncthreads();
#pragma unroll
  for (int r = 0; r < 16; ++r) {
    float o = acc[r] / nrm[c * 16 + r];
    int n = n0 + r;
    f32[((size_t)c * N + n) * KD + k] = o;
    f16[((size_t)c * N + n) * KD + k] = (_Float16)o;
    fT[((size_t)c * KD + k) * N + n] = (_Float16)o;
  }
}

// ---------------------------------------------------------------------------
// Kernel 3: one fused attention iteration.
// Block: 8 waves = 2 n-tiles (16 rows each) x 4 m-split waves.
// Wave: holds A-fragments for all 4 channels of its 16 rows; loops its m-range
// in 32-wide steps: 16 score WMMAs -> register channel-softmax + adj mask
// (adjacency via wave-uniform SMEM loads on KMcnt) -> P re-layout via per-wave
// LDS -> 16 aggregation WMMAs against featsT.  Tail: deterministic cross-wave
// combine in LDS, residual + L2-norm, output.
// ---------------------------------------------------------------------------
__global__ __launch_bounds__(256) void iterate_kernel(
    const unsigned* __restrict__ bitsT, float* __restrict__ f32,
    const _Float16* __restrict__ f16_in, const _Float16* __restrict__ fT_in,
    _Float16* __restrict__ f16_out, _Float16* __restrict__ fT_out,
    float* __restrict__ out) {
  extern __shared__ char smem[];
  float* aggL = (float*)smem;                    // 2 tiles * 4c * 16 * 64 f32 = 32KB
  _Float16* pL = (_Float16*)(smem + 8192 * 4);   // 8 waves * 4ch * 512 halves = 32KB
  const int tid = threadIdx.x;
  const int wave = tid >> 5, lane = tid & 31;
  const int tile = wave >> 2, wslot = wave & 3;
  const int n0 = blockIdx.x * 32 + tile * 16;
  const int l15 = lane & 15;
  const int hif = lane >> 4;
  const int hi8 = hif * 8;    // C-layout row offset
  const int klo = hif * 8;    // unused (A chunks start at koff)
  const int koff = hif * 16;  // B-fragment K base
  _Float16* pw = pL + wave * 2048;

  // Loop-invariant A fragments: 4 channels x (K0..31, K32..63)
  h16v A0[CH], A1[CH];
  {
    const int row = n0 + l15;
#pragma unroll
    for (int c = 0; c < CH; ++c) {
      const _Float16* rp = f16_in + ((size_t)c * N + row) * KD;
      union { h16v v; h8v h[2]; } u0, u1;
      u0.h[0] = *(const h8v*)(rp + koff);
      u0.h[1] = *(const h8v*)(rp + koff + 8);
      u1.h[0] = *(const h8v*)(rp + 32 + koff);
      u1.h[1] = *(const h8v*)(rp + 32 + koff + 8);
      A0[c] = u0.v;
      A1[c] = u1.v;
    }
  }
  f8v agg[CH][4];
  const f8v zf = {0.f, 0.f, 0.f, 0.f, 0.f, 0.f, 0.f, 0.f};
#pragma unroll
  for (int c = 0; c < CH; ++c)
#pragma unroll
    for (int q = 0; q < 4; ++q) agg[c][q] = zf;

  const int mBeg = wslot * (N / 4), mEnd = mBeg + (N / 4);
  for (int m0 = mBeg; m0 < mEnd; m0 += 32) {
    // --- adjacency: 16 words (rows n0..n0+15, one 32-bit word = m0..m0+31),
    // contiguous in adjT and wave-uniform -> scalar loads on KMcnt ---
    const int widx = __builtin_amdgcn_readfirstlane((m0 >> 5) * N + n0);
    const uint4* wp = (const uint4*)(bitsT + widx);
    const uint4 sa = wp[0], sb = wp[1], sc_ = wp[2], sd = wp[3];
    const unsigned lo[8] = {sa.x, sa.y, sa.z, sa.w, sb.x, sb.y, sb.z, sb.w};
    const unsigned hi[8] = {sc_.x, sc_.y, sc_.z, sc_.w, sd.x, sd.y, sd.z, sd.w};
    unsigned aw[8];
#pragma unroll
    for (int r = 0; r < 8; ++r) aw[r] = hif ? hi[r] : lo[r];

#pragma unroll
    for (int s = 0; s < 2; ++s) {
      // --- scores for 16x16 subtile (all 4 channels) ---
      f8v S[CH];
      const int mrow = m0 + s * 16 + l15;
#pragma unroll
      for (int c = 0; c < CH; ++c) {
        const _Float16* bp = f16_in + ((size_t)c * N + mrow) * KD;
        h16v B0 = *(const h16v*)(bp + koff);
        h16v B1 = *(const h16v*)(bp + 32 + koff);
        S[c] = wmma_f16(A1[c], B1, wmma_f16(A0[c], B0, zf));
      }
      // --- mask + channel softmax, elementwise across the 4 fragments ---
#pragma unroll
      for (int j = 0; j < 8; ++j) {
        float s0 = S[0][j], s1 = S[1][j], s2 = S[2][j], s3 = S[3][j];
        float mx = fmaxf(fmaxf(s0, s1), fmaxf(s2, s3));
        float e0 = __expf(s0 - mx), e1 = __expf(s1 - mx);
        float e2 = __expf(s2 - mx), e3 = __expf(s3 - mx);
        unsigned bit = (aw[j] >> (s * 16 + l15)) & 1u;
        float f = bit ? __builtin_amdgcn_rcpf(e0 + e1 + e2 + e3) : 0.0f;
        const int off = (j + hi8) * 32 + s * 16 + l15;  // row-major P in LDS
        pw[off] = (_Float16)(e0 * f);
        pw[512 + off] = (_Float16)(e1 * f);
        pw[1024 + off] = (_Float16)(e2 * f);
        pw[1536 + off] = (_Float16)(e3 * f);
      }
    }
    // --- aggregation: agg[c] += P_c(16x32) x featsT(32x16) per 16-k slice ---
    // (same-wave DS ops are hardware in-order: stores above complete first)
#pragma unroll
    for (int c = 0; c < CH; ++c) {
      const _Float16* pp = pw + c * 512 + l15 * 32;
      union { h16v v; h8v h[2]; } up;
      up.h[0] = *(const h8v*)(pp + koff);      // m-offsets koff..koff+7
      up.h[1] = *(const h8v*)(pp + koff + 8);  // m-offsets koff+8..koff+15
#pragma unroll
      for (int q = 0; q < 4; ++q) {
        const _Float16* tp =
            fT_in + ((size_t)c * KD + q * 16 + l15) * N + m0 + koff;
        h16v Bt = *(const h16v*)tp;
        agg[c][q] = wmma_f16(up.v, Bt, agg[c][q]);
      }
    }
  }

  // Deterministic cross-wave combine (round-robin, no float atomics).
  for (int w = 0; w < 4; ++w) {
    if (wslot == w) {
#pragma unroll
      for (int c = 0; c < CH; ++c)
#pragma unroll
        for (int q = 0; q < 4; ++q)
#pragma unroll
          for (int j = 0; j < 8; ++j) {
            const int idx = tile * 4096 + c * 1024 + (j + hi8) * 64 + q * 16 + l15;
            if (w == 0) aggL[idx] = agg[c][q][j];
            else        aggL[idx] += agg[c][q][j];
          }
    }
    __syncthreads();
  }

  // Finalize: residual + L2-norm; refresh f32/f16/fT (and output on last iter).
  if (tid < 128) {
    const int c = tid >> 5, row = tid & 31;
    const int t2 = row >> 4, r16 = row & 15;
    const int n = blockIdx.x * 32 + row;
    const float* ar = aggL + t2 * 4096 + c * 1024 + r16 * 64;
    float* fr = f32 + ((size_t)c * N + n) * KD;
    float ss = 0.f;
    for (int k = 0; k < KD; ++k) {
      float v = fr[k] + ar[k];
      ss += v * v;
    }
    const float inv = 1.0f / fmaxf(sqrtf(ss), 1e-12f);
    for (int k = 0; k < KD; ++k) {
      float o = (fr[k] + ar[k]) * inv;
      fr[k] = o;
      f16_out[((size_t)c * N + n) * KD + k] = (_Float16)o;
      fT_out[((size_t)c * KD + k) * N + n] = (_Float16)o;
      if (out) out[(size_t)n * (CH * KD) + c * KD + k] = o;
    }
  }
}

// ---------------------------------------------------------------------------
extern "C" void kernel_launch(void* const* d_in, const int* in_sizes, int n_in,
                              void* d_out, int out_size, void* d_ws,
                              size_t ws_size, hipStream_t stream) {
  const float* features = (const float*)d_in[0];
  const int* adj = (const int*)d_in[1];
  const float* W = (const float*)d_in[2];
  const float* b = (const float*)d_in[3];

  char* ws = (char*)d_ws;
  size_t o = 0;
  unsigned* bitsBuf = (unsigned*)(ws + o); o += (size_t)N * NW * 4;        // 4.5 MB
  float* f32 = (float*)(ws + o);           o += (size_t)CH * N * KD * 4;   // 6.0 MB
  _Float16* f16a = (_Float16*)(ws + o);    o += (size_t)CH * N * KD * 2;   // 3.0 MB
  _Float16* f16b = (_Float16*)(ws + o);    o += (size_t)CH * N * KD * 2;
  _Float16* fTa = (_Float16*)(ws + o);     o += (size_t)CH * N * KD * 2;
  _Float16* fTb = (_Float16*)(ws + o);     o += (size_t)CH * N * KD * 2;   // ~22.5 MB total

  pack_adj_kernel<<<(N * NW + 255) / 256, 256, 0, stream>>>(adj, bitsBuf);

  const size_t shProj = 16 * IN_DIM * 4 + CH * 16 * KD * 4 + 64 * 4;
  project_kernel<<<N / 16, 256, shProj, stream>>>(features, W, b, f32, f16a, fTa);

  const size_t shIter = 8192 * 4 + 8 * 2048 * 2;  // 64 KB
  iterate_kernel<<<N / 32, 256, shIter, stream>>>(bitsBuf, f32, f16a, fTa, f16b,
                                                  fTb, nullptr);
  iterate_kernel<<<N / 32, 256, shIter, stream>>>(bitsBuf, f32, f16b, fTb, f16a,
                                                  fTa, nullptr);
  iterate_kernel<<<N / 32, 256, shIter, stream>>>(bitsBuf, f32, f16a, fTa, f16b,
                                                  fTb, (float*)d_out);
}
